// ResidualVQ_6640019440247
// MI455X (gfx1250) — matrix operand order repaired and checked by reading, not compiled, not run on hardware
//
#include <hip/hip_runtime.h>
#include <hip/hip_bf16.h>

typedef float  v2f   __attribute__((ext_vector_type(2)));
typedef float  v8f   __attribute__((ext_vector_type(8)));
typedef __bf16 v16bf __attribute__((ext_vector_type(16)));
typedef __bf16 bf16_t;

#define N_ROWS 32768   // B*S = 8*4096
#define DIM    512
#define KCODES 4096
#define NLAYER 3
#define WAVES  8
#define PAD    8       // bf16 elements of row padding in LDS (keeps 16B alignment)

union BFrag { v16bf v; float4 f4[2]; };

// ---------------------------------------------------------------------------
// init: residual = z, qsum = 0
__global__ __launch_bounds__(256) void init_copy_kernel(const float* __restrict__ z,
                                                        float* __restrict__ residual,
                                                        float* __restrict__ qsum) {
    size_t i = (size_t)blockIdx.x * blockDim.x + threadIdx.x;  // float4 index
    float4 v = ((const float4*)z)[i];
    ((float4*)residual)[i] = v;
    ((float4*)qsum)[i] = make_float4(0.f, 0.f, 0.f, 0.f);
}

__global__ __launch_bounds__(256) void zero_misc_kernel(int* __restrict__ counts,
                                                        float* __restrict__ sumdist) {
    for (int i = threadIdx.x; i < NLAYER * KCODES; i += 256) counts[i] = 0;
    if (threadIdx.x < NLAYER) sumdist[threadIdx.x] = 0.f;
}

// ---------------------------------------------------------------------------
// split codebook into bf16 hi/lo planes:  E ~= Eh + El  (bf16x3 GEMM operands)
__global__ __launch_bounds__(256) void split_codebook_kernel(const float* __restrict__ E,
                                                             bf16_t* __restrict__ Eh,
                                                             bf16_t* __restrict__ El) {
    size_t i = (size_t)blockIdx.x * 256 + threadIdx.x;
    float x = E[i];
    bf16_t h = (bf16_t)x;
    Eh[i] = h;
    El[i] = (bf16_t)(x - (float)h);
}

// ---------------------------------------------------------------------------
// ||E_k||^2 per code row; one wave per row (from fp32 codebook: exact)
__global__ __launch_bounds__(256) void enorm_kernel(const float* __restrict__ E,
                                                    float* __restrict__ Enorm) {
    int k = blockIdx.x * WAVES + (threadIdx.x >> 5);
    int lane = threadIdx.x & 31;
    const float* row = E + (size_t)k * DIM;
    float s = 0.f;
    for (int c = lane; c < DIM; c += 32) {
        float x = row[c];
        s = __builtin_fmaf(x, x, s);
    }
#pragma unroll
    for (int m = 1; m < 32; m <<= 1) s += __shfl_xor(s, m, 32);
    if (lane == 0) Enorm[k] = s;
}

// ---------------------------------------------------------------------------
// Fused bf16x3 WMMA distance + argmin kernel.
// Block = 256 threads (8 waves), owns rows [16*blockIdx, 16*blockIdx+16).
// Each wave sweeps KCODES/WAVES codes in 16-column tiles; the 16x16 dot tile is
// accumulated with v_wmma_f32_16x16x32_bf16: 3 wmma per K=32 (hh + hl + lh).
__global__ __launch_bounds__(256) void vq_argmin_kernel(const float* __restrict__ R,
                                                        const bf16_t* __restrict__ Eh,
                                                        const bf16_t* __restrict__ El,
                                                        const float* __restrict__ Enorm,
                                                        int*   __restrict__ idx_out,
                                                        float* __restrict__ out_indices,
                                                        float* __restrict__ sumdist,
                                                        int*   __restrict__ counts,
                                                        int layer) {
    __shared__ bf16_t Ah[16][DIM + PAD];
    __shared__ bf16_t Al[16][DIM + PAD];
    __shared__ float rval[WAVES][16];
    __shared__ int   ridx[WAVES][16];
    __shared__ float rnorm[16];

    const int tid   = threadIdx.x;
    const int wave  = tid >> 5;
    const int lane  = tid & 31;
    const int n     = lane & 15;   // column within tile / row M for A fragment
    const int khalf = lane >> 4;   // K-half select within fragments

    const float* Rtile = R + (size_t)blockIdx.x * 16 * DIM;

    // stage A tile (16 x 512) as bf16 hi/lo planes
    for (int i = tid; i < 16 * (DIM / 4); i += 256) {
        int r  = i >> 7;            // / (DIM/4)
        int c4 = (i & 127) << 2;
        float4 v = *(const float4*)(Rtile + r * DIM + c4);
        float f[4] = {v.x, v.y, v.z, v.w};
#pragma unroll
        for (int u = 0; u < 4; ++u) {
            bf16_t h = (bf16_t)f[u];
            Ah[r][c4 + u] = h;
            Al[r][c4 + u] = (bf16_t)(f[u] - (float)h);
        }
    }
    __syncthreads();

    if (tid < 16) {  // row norms from reconstructed values (cheap vs wmma loop)
        float s = 0.f;
        for (int c = 0; c < DIM; ++c) {
            float x = (float)Ah[tid][c] + (float)Al[tid][c];
            s = __builtin_fmaf(x, x, s);
        }
        rnorm[tid] = s;
    }

    float best[8];
    int   bidx[8];
#pragma unroll
    for (int j = 0; j < 8; ++j) { best[j] = 3.4e38f; bidx[j] = 0x7fffffff; }

    for (int t = 0; t < KCODES / (WAVES * 16); ++t) {
        const int colbase = (t * WAVES + wave) * 16;   // waves cover adjacent tiles
        const bf16_t* ErH = Eh + (size_t)(colbase + n) * DIM + 16 * khalf;
        const bf16_t* ErL = El + (size_t)(colbase + n) * DIM + 16 * khalf;

        v8f acc = {};
#pragma unroll 4
        for (int c = 0; c < DIM; c += 32) {
            const int ao = c + 16 * khalf;
            BFrag ah, al, bh, bl;
            // A 16x32 bf16: v0-7 = K[c+16h .. +15]
            ah.f4[0] = *(const float4*)&Ah[n][ao];
            ah.f4[1] = *(const float4*)&Ah[n][ao + 8];
            al.f4[0] = *(const float4*)&Al[n][ao];
            al.f4[1] = *(const float4*)&Al[n][ao + 8];
            // B 32x16 bf16: lane(n,h) holds 16 consecutive K along the code row
            bh.f4[0] = *(const float4*)(ErH + c);
            bh.f4[1] = *(const float4*)(ErH + c + 8);
            bl.f4[0] = *(const float4*)(ErL + c);
            bl.f4[1] = *(const float4*)(ErL + c + 8);

            acc = __builtin_amdgcn_wmma_f32_16x16x32_bf16(
                false, ah.v, false, bh.v, (short)0, acc, false, false);
            acc = __builtin_amdgcn_wmma_f32_16x16x32_bf16(
                false, ah.v, false, bl.v, (short)0, acc, false, false);
            acc = __builtin_amdgcn_wmma_f32_16x16x32_bf16(
                false, al.v, false, bh.v, (short)0, acc, false, false);
        }

        const float en = Enorm[colbase + n];
        const int   ci = colbase + n;
#pragma unroll
        for (int j = 0; j < 8; ++j) {
            float d = __builtin_fmaf(-2.0f, acc[j], en);  // ||E||^2 - 2 z.E
            if (d < best[j] || (d == best[j] && ci < bidx[j])) { best[j] = d; bidx[j] = ci; }
        }
    }

    // butterfly min over the 16 lanes of each half (columns), per accumulator row
#pragma unroll
    for (int j = 0; j < 8; ++j) {
        float v = best[j]; int i = bidx[j];
#pragma unroll
        for (int m = 1; m < 16; m <<= 1) {
            float ov = __shfl_xor(v, m, 32);
            int   oi = __shfl_xor(i, m, 32);
            if (ov < v || (ov == v && oi < i)) { v = ov; i = oi; }
        }
        if (n == 0) {  // row = j + 8*khalf  (C/D layout: vgpr j, lane half)
            rval[wave][j + 8 * khalf] = v;
            ridx[wave][j + 8 * khalf] = i;
        }
    }
    __syncthreads();

    if (tid < 16) {  // cross-wave reduce, one thread per row
        float v = rval[0][tid]; int i = ridx[0][tid];
#pragma unroll
        for (int w = 1; w < WAVES; ++w) {
            float ov = rval[w][tid]; int oi = ridx[w][tid];
            if (ov < v || (ov == v && oi < i)) { v = ov; i = oi; }
        }
        int rowg = blockIdx.x * 16 + tid;
        idx_out[rowg] = i;
        out_indices[(size_t)rowg * NLAYER + layer] = (float)i;
        atomicAdd(&counts[i], 1);
        float dist = v + rnorm[tid];           // ~||z - E[idx]||^2
#pragma unroll
        for (int m = 1; m < 16; m <<= 1) dist += __shfl_xor(dist, m, 32);
        if (tid == 0) atomicAdd(sumdist, dist);
    }
}

// ---------------------------------------------------------------------------
// qsum += E[idx];  residual -= E[idx]   (fp32 codebook: exact ST output)
__global__ __launch_bounds__(256) void update_kernel(const float* __restrict__ E,
                                                     const int* __restrict__ idx,
                                                     float* __restrict__ residual,
                                                     float* __restrict__ qsum) {
    int row = blockIdx.x;
    int k = idx[row];
    const float2* q = (const float2*)(E + (size_t)k * DIM);
    float2* r = (float2*)(residual + (size_t)row * DIM);
    float2* s = (float2*)(qsum + (size_t)row * DIM);
    int t = threadIdx.x;
    float2 qv = q[t], rv = r[t], sv = s[t];
    rv.x -= qv.x; rv.y -= qv.y;
    sv.x += qv.x; sv.y += qv.y;
    r[t] = rv; s[t] = sv;
}

// ---------------------------------------------------------------------------
// commit + avg perplexity scalars
__global__ __launch_bounds__(256) void finalize_kernel(const int* __restrict__ counts,
                                                       const float* __restrict__ sumdist,
                                                       float* __restrict__ out_scalars) {
    __shared__ float sh[256];
    __shared__ float perp_acc;
    if (threadIdx.x == 0) perp_acc = 0.f;
    __syncthreads();
    const float invN = 1.0f / (float)N_ROWS;
    for (int l = 0; l < NLAYER; ++l) {
        float s = 0.f;
        for (int k = threadIdx.x; k < KCODES; k += 256) {
            float p = (float)counts[l * KCODES + k] * invN;
            s += p * logf(p + 1e-10f);
        }
        sh[threadIdx.x] = s;
        __syncthreads();
        for (int st = 128; st > 0; st >>= 1) {
            if (threadIdx.x < st) sh[threadIdx.x] += sh[threadIdx.x + st];
            __syncthreads();
        }
        if (threadIdx.x == 0) perp_acc += expf(-sh[0]);
        __syncthreads();
    }
    if (threadIdx.x == 0) {
        float commit = 0.25f * (sumdist[0] + sumdist[1] + sumdist[2]) /
                       ((float)N_ROWS * (float)DIM);
        out_scalars[0] = commit;
        out_scalars[1] = perp_acc / (float)NLAYER;
    }
}

// ---------------------------------------------------------------------------
extern "C" void kernel_launch(void* const* d_in, const int* in_sizes, int n_in,
                              void* d_out, int out_size, void* d_ws, size_t ws_size,
                              hipStream_t stream) {
    const float* z         = (const float*)d_in[0];
    const float* codebooks = (const float*)d_in[1];
    float* out = (float*)d_out;

    // workspace layout (16B-aligned sections first)
    float*  residual = (float*)d_ws;                             // N*D f32 (64 MB)
    bf16_t* Eh       = (bf16_t*)(residual + (size_t)N_ROWS * DIM);  // K*D bf16 (4 MB)
    bf16_t* El       = Eh + (size_t)KCODES * DIM;                // K*D bf16 (4 MB)
    float*  Enorm    = (float*)(El + (size_t)KCODES * DIM);      // K f32
    int*    idxbuf   = (int*)(Enorm + KCODES);                   // N i32
    int*    counts   = idxbuf + N_ROWS;                          // L*K i32
    float*  sumdist  = (float*)(counts + NLAYER * KCODES);       // L f32

    // output layout (flat, return order)
    float* qsum        = out;                                    // [B,S,D]
    float* out_idx     = out + (size_t)N_ROWS * DIM;             // [B,S,L] as float
    float* out_scalars = out_idx + (size_t)N_ROWS * NLAYER;      // commit, avg_perp

    init_copy_kernel<<<(N_ROWS * DIM / 4) / 256, 256, 0, stream>>>(z, residual, qsum);
    zero_misc_kernel<<<1, 256, 0, stream>>>(counts, sumdist);

    for (int l = 0; l < NLAYER; ++l) {
        const float* E = codebooks + (size_t)l * KCODES * DIM;
        split_codebook_kernel<<<(KCODES * DIM) / 256, 256, 0, stream>>>(E, Eh, El);
        enorm_kernel<<<KCODES / WAVES, 256, 0, stream>>>(E, Enorm);
        vq_argmin_kernel<<<N_ROWS / 16, 256, 0, stream>>>(
            residual, Eh, El, Enorm, idxbuf, out_idx, sumdist + l, counts + l * KCODES, l);
        update_kernel<<<N_ROWS, 256, 0, stream>>>(E, idxbuf, residual, qsum);
    }
    finalize_kernel<<<1, 256, 0, stream>>>(counts, sumdist, out_scalars);
}
